// OkutamaNet_34394098107057
// MI455X (gfx1250) — matrix operand order repaired and checked by reading, not compiled, not run on hardware
//
#include <hip/hip_runtime.h>
#include <stdint.h>

#define DIN 26400
#define NFB 512
#define NACT 13
#define MAXN 12
#define BDIM 8
#define TDIM 20
#define BT (BDIM * TDIM)        /* 160  */
#define MROWS (BT * MAXN)       /* 1920 */
#define KSTEPS 825              /* DIN/32 */
#define KSPLIT0 416             /* even step count for chunk 0 */
#define LNEPS 1e-5f
#define LPAD 516                /* h_lds row stride, multiple of 4 */

typedef __attribute__((ext_vector_type(16))) __bf16 v16bf;
typedef __attribute__((ext_vector_type(16))) float  v16f;
typedef __attribute__((ext_vector_type(8)))  float  v8f;

__device__ __forceinline__ unsigned short f2bf(float f) {
  union { float f; unsigned int u; } c; c.f = f;
  unsigned int u = c.u;
  return (unsigned short)((u + 0x7fffu + ((u >> 16) & 1u)) >> 16);  // RNE
}

// A fragment: 16 fp32 -> v16bf via native pack-convert
__device__ __forceinline__ v16bf loadA(const float* p, int k) {
  float4 x0 = *(const float4*)(p + k);
  float4 x1 = *(const float4*)(p + k + 4);
  float4 x2 = *(const float4*)(p + k + 16);
  float4 x3 = *(const float4*)(p + k + 20);
  v16f t = { x0.x, x0.y, x0.z, x0.w, x1.x, x1.y, x1.z, x1.w,
             x2.x, x2.y, x2.z, x2.w, x3.x, x3.y, x3.z, x3.w };
  return __builtin_convertvector(t, v16bf);
}

__device__ __forceinline__ v16bf loadB16(const unsigned short* p) {
  union { v16bf v; uint4 q[2]; } u;
  u.q[0] = *(const uint4*)(p);
  u.q[1] = *(const uint4*)(p + 16);
  return u.v;
}

#define WMMA_BF16(A, B, C)                                                   \
  __builtin_amdgcn_wmma_f32_16x16x32_bf16(false, (A), false, (B), (short)0,  \
                                          (C), false, false)

// ---------------------------------------------------------------------------
// Kernel 1: W_emb fp32 -> bf16 (L2-resident afterwards)
// ---------------------------------------------------------------------------
__global__ void convert_w_kernel(const float* __restrict__ W,
                                 unsigned short* __restrict__ Wb, int n) {
  int i = (blockIdx.x * blockDim.x + threadIdx.x) * 4;
  if (i + 3 < n) {
    float4 v = *(const float4*)(W + i);
    Wb[i + 0] = f2bf(v.x); Wb[i + 1] = f2bf(v.y);
    Wb[i + 2] = f2bf(v.z); Wb[i + 3] = f2bf(v.w);
  }
}

// ---------------------------------------------------------------------------
// Kernel 2: bf16 WMMA GEMM, 32-row M-tiles, K split in two partial buffers.
// Grid (60, 2), 256 threads = 8 waves; wave owns 32 rows x 64 cols
// (2 M-subtiles x 4 N-subtiles = 8 WMMA accumulators).
// Depth-1 register double buffering: prefetch k+32 fragments, WMMA k.
// ---------------------------------------------------------------------------
__global__ __launch_bounds__(256)
void gemm_kernel(const float* __restrict__ X,            // [1920, 26400]
                 const unsigned short* __restrict__ Wb,  // [512, 26400] bf16
                 float* __restrict__ hpart)              // [2][1920][512]
{
  const int tid  = threadIdx.x;
  const int lane = tid & 31;
  const int wv   = tid >> 5;              // 0..7
  const int m0   = blockIdx.x * 32;
  const int ky   = blockIdx.y;            // 0..1
  const int kbeg = (ky == 0) ? 0 : KSPLIT0 * 32;
  const int steps = (ky == 0) ? KSPLIT0 : (KSTEPS - KSPLIT0);

  // A addressing (ISA 7.12.2 16-bit A layout)
  const int kAoff = (lane >> 4) * 8;
  const float* xr0 = X + (size_t)(m0 + (lane & 15)) * DIN + kAoff;
  const float* xr1 = xr0 + (size_t)16 * DIN;
  // B addressing: lane(0..15) holds column f, K-contiguous in memory
  const unsigned short* wb =
      Wb + (size_t)(wv * 64 + (lane & 15)) * DIN + (lane >> 4) * 8;

  v8f acc[8];
#pragma unroll
  for (int i = 0; i < 8; ++i) acc[i] = (v8f){};

  int kc = kbeg;
  v16bf a0 = loadA(xr0, kc), a1 = loadA(xr1, kc);
  v16bf b0 = loadB16(wb + kc);
  v16bf b1 = loadB16(wb + kc + (size_t)(16) * DIN);
  v16bf b2 = loadB16(wb + kc + (size_t)(32) * DIN);
  v16bf b3 = loadB16(wb + kc + (size_t)(48) * DIN);

#pragma unroll 2
  for (int s = 0; s < steps; ++s) {
    const int kn = (s + 1 < steps) ? (kc + 32) : kbeg;  // clamped dummy at end
    __builtin_prefetch((const void*)(xr0 + kn + 128), 0, 1);
    // prefetch next step's fragments into fresh registers
    v16bf na0 = loadA(xr0, kn), na1 = loadA(xr1, kn);
    v16bf nb0 = loadB16(wb + kn);
    v16bf nb1 = loadB16(wb + kn + (size_t)(16) * DIN);
    v16bf nb2 = loadB16(wb + kn + (size_t)(32) * DIN);
    v16bf nb3 = loadB16(wb + kn + (size_t)(48) * DIN);
    // 8 WMMAs on the current fragments
    acc[0] = WMMA_BF16(a0, b0, acc[0]);
    acc[1] = WMMA_BF16(a0, b1, acc[1]);
    acc[2] = WMMA_BF16(a0, b2, acc[2]);
    acc[3] = WMMA_BF16(a0, b3, acc[3]);
    acc[4] = WMMA_BF16(a1, b0, acc[4]);
    acc[5] = WMMA_BF16(a1, b1, acc[5]);
    acc[6] = WMMA_BF16(a1, b2, acc[6]);
    acc[7] = WMMA_BF16(a1, b3, acc[7]);
    a0 = na0; a1 = na1; b0 = nb0; b1 = nb1; b2 = nb2; b3 = nb3;
    kc = kn;
  }

  // write fp32 partials (C layout: lane<16 -> M rows 0..7, N=lane)
  float* hp = hpart + (size_t)ky * MROWS * NFB;
#pragma unroll
  for (int mi = 0; mi < 2; ++mi) {
#pragma unroll
    for (int t = 0; t < 4; ++t) {
      const int col = wv * 64 + t * 16 + (lane & 15);
      const int rb  = m0 + mi * 16 + (lane >> 4) * 8;
      const v8f a = acc[mi * 4 + t];
#pragma unroll
      for (int j = 0; j < 8; ++j)
        hp[(size_t)(rb + j) * NFB + col] = a[j];
    }
  }
}

// ---------------------------------------------------------------------------
// Kernel 3: sum partials + bias + relu + LayerNorm + 512->13 + argmax/labels
// 120 WGs x 16 rows, 256 threads.
// ---------------------------------------------------------------------------
__global__ __launch_bounds__(256)
void epilogue_kernel(const float* __restrict__ hpart,    // [2][1920][512]
                     const float* __restrict__ b_emb,
                     const float* __restrict__ ln_g,
                     const float* __restrict__ ln_b,
                     const float* __restrict__ W_act,    // [13,512]
                     const float* __restrict__ b_act,
                     const int*   __restrict__ nbox,     // [160]
                     int*   __restrict__ labels,         // [1920]
                     float* __restrict__ out_actn)       // [12*13] @ d_out
{
  __shared__ float h_lds[16][LPAD];
  __shared__ float mu_s[16], rs_s[16];
  __shared__ float sc_s[16][16];

  const int tid = threadIdx.x;
  const int m0  = blockIdx.x * 16;
  const int row = tid >> 4, seg = tid & 15;
  const int gm  = m0 + row;

  // sum K-partials + bias + relu -> LDS
  {
    const float* p0 = hpart + (size_t)gm * NFB;
    const float* p1 = p0 + (size_t)MROWS * NFB;
#pragma unroll
    for (int j = 0; j < 32; j += 4) {
      int c = seg * 32 + j;
      float4 u = *(const float4*)(p0 + c);
      float4 v = *(const float4*)(p1 + c);
      float4 b = *(const float4*)(b_emb + c);
      float r0 = u.x + v.x + b.x, r1 = u.y + v.y + b.y;
      float r2 = u.z + v.z + b.z, r3 = u.w + v.w + b.w;
      h_lds[row][c + 0] = r0 > 0.f ? r0 : 0.f;
      h_lds[row][c + 1] = r1 > 0.f ? r1 : 0.f;
      h_lds[row][c + 2] = r2 > 0.f ? r2 : 0.f;
      h_lds[row][c + 3] = r3 > 0.f ? r3 : 0.f;
    }
  }
  __syncthreads();

  // LayerNorm stats (one thread per row; 512 elems, tiny)
  if (tid < 16) {
    float s = 0.f, ss = 0.f;
    const float* hr = &h_lds[tid][0];
    for (int c = 0; c < NFB; c += 4) {
      float4 v = *(const float4*)(hr + c);
      s  += v.x + v.y + v.z + v.w;
      ss += v.x * v.x + v.y * v.y + v.z * v.z + v.w * v.w;
    }
    float mu  = s * (1.f / NFB);
    float var = ss * (1.f / NFB) - mu * mu;
    mu_s[tid] = mu;
    rs_s[tid] = rsqrtf(var + LNEPS);
  }
  __syncthreads();

  // normalize in place
  {
    float mu = mu_s[row], rs = rs_s[row];
#pragma unroll 8
    for (int j = 0; j < 32; ++j) {
      int c = seg * 32 + j;
      float v = h_lds[row][c];
      h_lds[row][c] = (v - mu) * rs * ln_g[c] + ln_b[c];
    }
  }
  __syncthreads();

  // scores: 16 rows x 13 actions
  if (tid < 16 * NACT) {
    int r = tid / NACT, a = tid % NACT;
    const float* wa = W_act + a * NFB;
    float d = b_act[a];
    for (int f = 0; f < NFB; ++f) d += h_lds[r][f] * wa[f];
    sc_s[r][a] = d;
    int g = m0 + r;
    if (g >= MROWS - MAXN)   // rows of the last (b,t): actn_score output
      out_actn[(g - (MROWS - MAXN)) * NACT + a] = d;
  }
  __syncthreads();

  // argmax (ties -> smallest) + valid mask -> labels
  if (tid < 16) {
    int g = m0 + tid;
    int bt = g / MAXN, n = g % MAXN;
    int N = nbox[bt]; N = N < 0 ? 0 : (N > MAXN ? MAXN : N);
    float best = sc_s[tid][0]; int bi = 0;
#pragma unroll
    for (int a = 1; a < NACT; ++a) {
      float v = sc_s[tid][a];
      if (v > best) { best = v; bi = a; }
    }
    labels[g] = (n < N) ? bi : 0;
  }
}

// ---------------------------------------------------------------------------
// Kernel 4: torch.mode over T (ties -> smallest class), 96 outputs
// ---------------------------------------------------------------------------
__global__ void mode_kernel(const int* __restrict__ labels,
                            float* __restrict__ out) {
  int idx = threadIdx.x;
  if (idx >= BDIM * MAXN) return;
  int b = idx / MAXN, n = idx % MAXN;
  int cnt[NACT];
#pragma unroll
  for (int a = 0; a < NACT; ++a) cnt[a] = 0;
  for (int t = 0; t < TDIM; ++t)
    cnt[labels[(b * TDIM + t) * MAXN + n]]++;
  int best = cnt[0], bi = 0;
#pragma unroll
  for (int a = 1; a < NACT; ++a)
    if (cnt[a] > best) { best = cnt[a]; bi = a; }
  out[idx] = (float)bi;
}

// ---------------------------------------------------------------------------
extern "C" void kernel_launch(void* const* d_in, const int* in_sizes, int n_in,
                              void* d_out, int out_size, void* d_ws, size_t ws_size,
                              hipStream_t stream) {
  (void)in_sizes; (void)n_in; (void)out_size; (void)ws_size;
  const float* X     = (const float*)d_in[1];   // boxes_features_all
  const int*   nbox  = (const int*)  d_in[2];   // bboxes_num_in
  const float* W_emb = (const float*)d_in[4];
  const float* b_emb = (const float*)d_in[5];
  const float* ln_g  = (const float*)d_in[6];
  const float* ln_b  = (const float*)d_in[7];
  const float* W_act = (const float*)d_in[8];
  const float* b_act = (const float*)d_in[9];
  float* out = (float*)d_out;

  unsigned short* Wb = (unsigned short*)d_ws;                 // 27 MB bf16 W
  float* hpart = (float*)((char*)d_ws + (size_t)NFB * DIN * 2);      // 7.9 MB
  int* labels  = (int*)((char*)hpart + (size_t)2 * MROWS * NFB * 4); // 7.7 KB

  const int nW = NFB * DIN;                                   // 13,516,800
  convert_w_kernel<<<(nW / 4 + 255) / 256, 256, 0, stream>>>(W_emb, Wb, nW);
  gemm_kernel<<<dim3(MROWS / 32, 2), 256, 0, stream>>>(X, Wb, hpart);
  epilogue_kernel<<<MROWS / 16, 256, 0, stream>>>(
      hpart, b_emb, ln_g, ln_b, W_act, b_act, nbox, labels, out);
  mode_kernel<<<1, 128, 0, stream>>>(labels, out + MAXN * NACT);
}
